// AttnModel_64098091926054
// MI455X (gfx1250) — compile-verified, matrix operand order checked by reading, not run on hardware
//
#include <hip/hip_runtime.h>

// ---------------------------------------------------------------------------
// Model constants
// ---------------------------------------------------------------------------
constexpr int kBS    = 9;
constexpr int kP     = 81;
constexpr int kD     = 512;
constexpr int kB     = 256;
constexpr int kL     = 8;
constexpr int kDPREP = 19;
constexpr int kN1    = 1536;   // fused [kv(1024) | q(512)] output width

typedef __bf16 bf16_t;
typedef __attribute__((ext_vector_type(16))) __bf16 v16bf;
typedef __attribute__((ext_vector_type(8)))  float  v8f;

// Async global->LDS (CDNA5 ASYNCcnt path); guarded so the fallback is the
// proven manual global_load_b128 + ds_store_b128 copy.
#if __has_builtin(__builtin_amdgcn_global_load_async_to_lds_b128) && \
    __has_builtin(__builtin_amdgcn_s_wait_asynccnt)
#define USE_ASYNC_LDS 1
// clang prototype: void(v4i addrspace(1)* src, v4i addrspace(3)* dst,
//                       imm int offset, imm int cpol)
typedef int v4i_t __attribute__((vector_size(16)));
typedef __attribute__((address_space(1))) v4i_t* gas_v4i_p;   // global int4*
typedef __attribute__((address_space(3))) v4i_t* las_v4i_p;   // LDS int4*
#endif

// ---------------------------------------------------------------------------
// bf16 WMMA GEMM: C(M,N) = A(M,K) x Bt(N,K)^T (+ epilogue)
// A row-major (M x K), Bt row-major (N x K)  -> both tiles load identically.
// block = 128 threads (4 waves, wave32), block tile 64x64, K-step 32,
// double-buffered LDS (one barrier per K-step); each wave owns a 32x32
// region = 2x2 v_wmma_f32_16x16x32_bf16 tiles.
// EPI: 0 = store f32 (+bias)
//      1 = x += alpha*relu(acc+bias); store x (f32) and x16 (bf16)
//      2 = store bf16 relu(acc+bias)
//      3 = x += alpha*(acc+bias);     store x (f32) and x16 (bf16)
// ---------------------------------------------------------------------------
template <int EPI>
__global__ __launch_bounds__(128) void gemm_bf16_k(
    const bf16_t* __restrict__ A, const bf16_t* __restrict__ Bt,
    int M, int N, int K,
    const float* __restrict__ bias,
    float* __restrict__ outf,
    float* __restrict__ xbuf,
    bf16_t* __restrict__ out16,
    const float* __restrict__ alphap)
{
    // stride 40 halfs (80B) keeps every 8-half fragment read 16B-aligned
    __shared__ __align__(16) bf16_t As[2][64][40];
    __shared__ __align__(16) bf16_t Bs[2][64][40];

    const int m0   = blockIdx.x * 64;
    const int n0   = blockIdx.y * 64;
    const int tid  = threadIdx.x;
    const int lane = tid & 31;
    const int wid  = tid >> 5;
    const int wm   = (wid >> 1) * 32;             // wave row offset in tile
    const int wn   = (wid & 1) * 32;              // wave col offset in tile
    const int r    = lane & 15;
    const int h    = lane >> 4;

    // cooperative loader coordinates (128 threads, 16 halfs per matrix each)
    const int lrow = tid >> 1;                    // 0..63
    const int lcol = (tid & 1) * 16;              // 0 / 16

    const bf16_t* gA = A  + (size_t)(m0 + lrow) * K + lcol;
    const bf16_t* gB = Bt + (size_t)(n0 + lrow) * K + lcol;

    auto load_tiles = [&](int k0, int p) {
#ifdef USE_ASYNC_LDS
        // direct global->LDS DMA, no VGPR round trip; tracked by ASYNCcnt
        __builtin_amdgcn_global_load_async_to_lds_b128(
            (gas_v4i_p)(gA + k0),     (las_v4i_p)&As[p][lrow][lcol],     0, 0);
        __builtin_amdgcn_global_load_async_to_lds_b128(
            (gas_v4i_p)(gA + k0 + 8), (las_v4i_p)&As[p][lrow][lcol + 8], 0, 0);
        __builtin_amdgcn_global_load_async_to_lds_b128(
            (gas_v4i_p)(gB + k0),     (las_v4i_p)&Bs[p][lrow][lcol],     0, 0);
        __builtin_amdgcn_global_load_async_to_lds_b128(
            (gas_v4i_p)(gB + k0 + 8), (las_v4i_p)&Bs[p][lrow][lcol + 8], 0, 0);
#else
        const uint4* ga = (const uint4*)(gA + k0);
        *(uint4*)&As[p][lrow][lcol]     = ga[0];
        *(uint4*)&As[p][lrow][lcol + 8] = ga[1];
        const uint4* gb = (const uint4*)(gB + k0);
        *(uint4*)&Bs[p][lrow][lcol]     = gb[0];
        *(uint4*)&Bs[p][lrow][lcol + 8] = gb[1];
#endif
    };

    v8f acc[2][2] = {};

    load_tiles(0, 0);
#ifdef USE_ASYNC_LDS
    __builtin_amdgcn_s_wait_asynccnt(0);
#endif
    __syncthreads();

    int p = 0;
    for (int k0 = 0; k0 < K; k0 += 32) {
        if (k0 + 32 < K) {
            load_tiles(k0 + 32, p ^ 1);           // fill other buffer
            if (k0 + 64 < K) {                    // pull tile after that into caches
                __builtin_prefetch(gA + k0 + 64, 0, 3);
                __builtin_prefetch(gB + k0 + 64, 0, 3);
            }
        }

        // ---- fragments: per lane two 16B LDS reads, K runs {h*8.., 16+h*8..} ----
        v16bf af[2], bf_[2];
#pragma unroll
        for (int i = 0; i < 2; ++i) {
            union { v16bf v; uint4 q[2]; } ua;
            const bf16_t* pa = &As[p][wm + i * 16 + r][h * 8];
            ua.q[0] = *(const uint4*)pa;
            ua.q[1] = *(const uint4*)(pa + 16);
            af[i] = ua.v;
        }
#pragma unroll
        for (int j = 0; j < 2; ++j) {
            union { v16bf v; uint4 q[2]; } ub;
            const bf16_t* pb = &Bs[p][wn + j * 16 + r][h * 8];
            ub.q[0] = *(const uint4*)pb;
            ub.q[1] = *(const uint4*)(pb + 16);
            bf_[j] = ub.v;
        }

#pragma unroll
        for (int i = 0; i < 2; ++i)
#pragma unroll
            for (int j = 0; j < 2; ++j)
                acc[i][j] = __builtin_amdgcn_wmma_f32_16x16x32_bf16(
                    false, af[i], false, bf_[j], (short)0, acc[i][j], false, false);

#ifdef USE_ASYNC_LDS
        __builtin_amdgcn_s_wait_asynccnt(0);      // next buffer's DMA landed
#endif
        __syncthreads();                          // reads of p done, writes of p^1 done
        p ^= 1;
    }

    float alpha = 0.f;
    if constexpr (EPI == 1 || EPI == 3) alpha = *alphap;

    // C/D layout: element g of lane -> row = base + h*8 + g, col = base + r
#pragma unroll
    for (int i = 0; i < 2; ++i) {
#pragma unroll
        for (int j = 0; j < 2; ++j) {
            const int col = n0 + wn + j * 16 + r;
            const float bc = bias[col];
#pragma unroll
            for (int g = 0; g < 8; ++g) {
                const int row = m0 + wm + i * 16 + h * 8 + g;
                const size_t idx = (size_t)row * N + col;
                float v = acc[i][j][g] + bc;
                if constexpr (EPI == 0) {
                    outf[idx] = v;
                } else if constexpr (EPI == 1) {
                    v = fmaxf(v, 0.f);
                    const float xn = xbuf[idx] + alpha * v;
                    xbuf[idx]  = xn;
                    out16[idx] = (bf16_t)xn;
                } else if constexpr (EPI == 2) {
                    out16[idx] = (bf16_t)fmaxf(v, 0.f);
                } else {
                    const float xn = xbuf[idx] + alpha * v;
                    xbuf[idx]  = xn;
                    out16[idx] = (bf16_t)xn;
                }
            }
        }
    }
}

// ---------------------------------------------------------------------------
// LDS-tiled transpose + f32->bf16: src (R x C) row-major -> dst (C x R) bf16
// coalesced reads and writes; grid (C/32, R/32), block 256
// ---------------------------------------------------------------------------
__global__ __launch_bounds__(256) void tpose_bf16_k(const float* __restrict__ src,
                                                    bf16_t* __restrict__ dst,
                                                    int R, int C)
{
    __shared__ float tile[32][33];
    const int c0 = blockIdx.x * 32;
    const int r0 = blockIdx.y * 32;
    const int tx = threadIdx.x & 31;
    const int ty = threadIdx.x >> 5;   // 0..7
#pragma unroll
    for (int i = 0; i < 4; ++i) {
        const int rr = ty + i * 8;
        tile[rr][tx] = src[(size_t)(r0 + rr) * C + c0 + tx];
    }
    __syncthreads();
#pragma unroll
    for (int i = 0; i < 4; ++i) {
        const int cc = ty + i * 8;
        dst[(size_t)(c0 + cc) * R + r0 + tx] = (bf16_t)tile[tx][cc];
    }
}

// ---------------------------------------------------------------------------
// prepare: neighbourhood features + positional encodings  -> bfb (B,P,19)
// ---------------------------------------------------------------------------
__global__ void prep_kernel(const float* __restrict__ obs,
                            float* __restrict__ bfb, float* __restrict__ posb)
{
    const int p = threadIdx.x;
    const int b = blockIdx.x;
    if (p >= kP) return;
    const int i = p / kBS, j = p % kBS;

    float pos[12];
    {
        const float ri = (float)i / (kBS - 1);
        const float ci = (float)j / (kBS - 1);
        const float zi = 0.5f * (ri + ci);
        const float periods[2] = {4.0f / (kBS - 1), 16.0f / (kBS - 1)};
        int o = 0;
#pragma unroll
        for (int t = 0; t < 2; ++t) {
            const float w = 2.0f * 3.14159265358979323846f / periods[t];
            pos[o + 0] = cosf(w * ri); pos[o + 1] = cosf(w * ci); pos[o + 2] = cosf(w * zi);
            pos[o + 3] = sinf(w * ri); pos[o + 4] = sinf(w * ci); pos[o + 5] = sinf(w * zi);
            o += 6;
        }
    }
    if (b == 0)
        for (int c = 0; c < 12; ++c) posb[p * 12 + c] = pos[c];

    const int dr[7] = {-1, -1, 0, 0, 0, -1, -1};
    const int dc[7] = { 0,  1, -1, 0, 1, -1,  0};
    float* out = bfb + ((size_t)b * kP + p) * kDPREP;
#pragma unroll
    for (int k = 0; k < 7; ++k) {
        const int ri2 = i + dr[k], ci2 = j + dc[k];
        float v = 0.f;
        if (ri2 >= 0 && ri2 < kBS && ci2 >= 0 && ci2 < kBS) {
            const size_t base = (((size_t)b * kBS + ri2) * kBS + ci2) * 2;
            v = obs[base] - obs[base + 1];
        }
        out[k] = v;
    }
#pragma unroll
    for (int c = 0; c < 12; ++c) out[7 + c] = pos[c];
}

// ---------------------------------------------------------------------------
// s[b,j] = sum_d q[b,d] * kvb_w_k[j,d]   (rank-19 attention projection)
// ---------------------------------------------------------------------------
__global__ __launch_bounds__(64) void s_kernel(const float* __restrict__ out1,
                                               const float* __restrict__ kvbw_l,
                                               float* __restrict__ sbuf)
{
    const int j = blockIdx.x;      // 0..18
    const int b = blockIdx.y;      // 0..255
    const float* q = out1 + (size_t)b * kN1 + 1024;
    const float* w = kvbw_l + (size_t)j * 1024;   // k half = first 512
    float p = 0.f;
    for (int d = threadIdx.x; d < kD; d += 64) p += q[d] * w[d];
    for (int o = 16; o > 0; o >>= 1) p += __shfl_down(p, o, 32);
    __shared__ float tmp[2];
    if ((threadIdx.x & 31) == 0) tmp[threadIdx.x >> 5] = p;
    __syncthreads();
    if (threadIdx.x == 0) sbuf[b * 32 + j] = tmp[0] + tmp[1];
}

// ---------------------------------------------------------------------------
// attention: dots -> softmax(81) -> t (19) -> vals = kvx_v + t @ Wv
// ---------------------------------------------------------------------------
__global__ __launch_bounds__(128) void attn_kernel(
    const float* __restrict__ out1, const float* __restrict__ sbuf,
    const float* __restrict__ bfb, const float* __restrict__ kvbw_l,
    float* __restrict__ vals, bf16_t* __restrict__ vals16)
{
    __shared__ float sh_dots[kP];
    __shared__ float sh_s[kDPREP];
    __shared__ float sh_t[kDPREP];
    __shared__ float red[4];
    __shared__ float sc[2];

    const int b = blockIdx.x, tid = threadIdx.x;
    const float* row = out1 + (size_t)b * kN1;   // [kvx_k(512)|kvx_v(512)|q(512)]
    const float* q   = row + 1024;

    float pp = 0.f;                              // dot0 = q . kvx_k
    for (int d = tid; d < kD; d += 128) pp += q[d] * row[d];
    for (int o = 16; o > 0; o >>= 1) pp += __shfl_down(pp, o, 32);
    if ((tid & 31) == 0) red[tid >> 5] = pp;
    if (tid < kDPREP) sh_s[tid] = sbuf[b * 32 + tid];
    __syncthreads();
    if (tid == 0) sc[0] = red[0] + red[1] + red[2] + red[3];
    __syncthreads();

    const float scale = 0.044194173824159216f;   // 1/sqrt(512)
    if (tid < kP) {
        const float* bp = bfb + ((size_t)b * kP + tid) * kDPREP;
        float ds = sc[0];
#pragma unroll
        for (int j = 0; j < kDPREP; ++j) ds += bp[j] * sh_s[j];
        sh_dots[tid] = ds * scale;
    }
    __syncthreads();
    if (tid == 0) {
        float m = sh_dots[0];
        for (int p2 = 1; p2 < kP; ++p2) m = fmaxf(m, sh_dots[p2]);
        float ssum = 0.f;
        for (int p2 = 0; p2 < kP; ++p2) ssum += expf(sh_dots[p2] - m);
        sc[0] = m; sc[1] = 1.f / ssum;
    }
    __syncthreads();
    if (tid < kP) sh_dots[tid] = expf(sh_dots[tid] - sc[0]) * sc[1];
    __syncthreads();
    if (tid < kDPREP) {
        float t = 0.f;
        for (int p2 = 0; p2 < kP; ++p2)
            t += sh_dots[p2] * bfb[((size_t)b * kP + p2) * kDPREP + tid];
        sh_t[tid] = t;
    }
    __syncthreads();
    for (int d = tid; d < kD; d += 128) {
        float v = row[512 + d];                  // kvx_v (biases folded in)
#pragma unroll
        for (int j = 0; j < kDPREP; ++j) v += sh_t[j] * kvbw_l[(size_t)j * 1024 + 512 + d];
        vals[(size_t)b * kD + d]   = v;
        vals16[(size_t)b * kD + d] = (bf16_t)v;
    }
}

// ---------------------------------------------------------------------------
// head: log_softmax((x @ head_w) @ pos^T)
// ---------------------------------------------------------------------------
__global__ __launch_bounds__(128) void head_kernel(
    const float* __restrict__ x, const float* __restrict__ headw,
    const float* __restrict__ posb, float* __restrict__ out)
{
    __shared__ float u[12];
    __shared__ float lg[kP];
    __shared__ float sc[2];
    const int b = blockIdx.x, tid = threadIdx.x;
    if (tid < 12) {
        const float* xr = x + (size_t)b * kD;
        float a = 0.f;
        for (int d = 0; d < kD; ++d) a += xr[d] * headw[d * 12 + tid];
        u[tid] = a;
    }
    __syncthreads();
    if (tid < kP) {
        float a = 0.f;
#pragma unroll
        for (int j = 0; j < 12; ++j) a += u[j] * posb[tid * 12 + j];
        lg[tid] = a;
    }
    __syncthreads();
    if (tid == 0) {
        float m = lg[0];
        for (int p2 = 1; p2 < kP; ++p2) m = fmaxf(m, lg[p2]);
        float ssum = 0.f;
        for (int p2 = 0; p2 < kP; ++p2) ssum += expf(lg[p2] - m);
        sc[0] = m; sc[1] = logf(ssum);
    }
    __syncthreads();
    if (tid < kP) out[(size_t)b * kP + tid] = lg[tid] - sc[0] - sc[1];
}

// ---------------------------------------------------------------------------
// small utility kernels
// ---------------------------------------------------------------------------
__global__ void bias1_k(const float* __restrict__ kvxb, const float* __restrict__ kvbb,
                        const float* __restrict__ qb, float* __restrict__ bias1)
{
    const int i = blockIdx.x * blockDim.x + threadIdx.x;
    if (i >= kN1) return;
    bias1[i] = (i < 1024) ? (kvxb[i] + kvbb[i]) : qb[i - 1024];
}

__global__ void init_x_k(float* __restrict__ x, bf16_t* __restrict__ x16)
{
    const int i = blockIdx.x * blockDim.x + threadIdx.x;
    if (i < kB * kD) { x[i] = 0.f; x16[i] = (bf16_t)0.f; }
}

// ---------------------------------------------------------------------------
// launcher
// ---------------------------------------------------------------------------
extern "C" void kernel_launch(void* const* d_in, const int* in_sizes, int n_in,
                              void* d_out, int out_size, void* d_ws, size_t ws_size,
                              hipStream_t stream)
{
    const float* obs    = (const float*)d_in[0];
    const float* kvx_w  = (const float*)d_in[1];
    const float* kvx_b  = (const float*)d_in[2];
    const float* kvb_w  = (const float*)d_in[3];
    const float* kvb_b  = (const float*)d_in[4];
    const float* q_w    = (const float*)d_in[5];
    const float* q_b    = (const float*)d_in[6];
    const float* fin_w  = (const float*)d_in[7];
    const float* fin_b  = (const float*)d_in[8];
    const float* fc0_w  = (const float*)d_in[9];
    const float* fc0_b  = (const float*)d_in[10];
    const float* fc1_w  = (const float*)d_in[11];
    const float* fc1_b  = (const float*)d_in[12];
    const float* alpha  = (const float*)d_in[13];
    const float* head_w = (const float*)d_in[14];
    float* out = (float*)d_out;

    char* ws = (char*)d_ws;
    size_t off = 0;
    auto take = [&](size_t bytes) -> char* {
        char* p = ws + off;
        off += (bytes + 255) & ~(size_t)255;
        return p;
    };
    float*  x      = (float*) take((size_t)kB * kD * 4);
    bf16_t* x16    = (bf16_t*)take((size_t)kB * kD * 2);
    float*  out1   = (float*) take((size_t)kB * kN1 * 4);
    float*  sbuf   = (float*) take((size_t)kB * 32 * 4);
    float*  bfb    = (float*) take((size_t)kB * kP * kDPREP * 4);
    float*  posb   = (float*) take((size_t)kP * 12 * 4);
    float*  valsf  = (float*) take((size_t)kB * kD * 4);
    bf16_t* vals16 = (bf16_t*)take((size_t)kB * kD * 2);
    bf16_t* h16    = (bf16_t*)take((size_t)kB * kD * 2);
    float*  bias1  = (float*) take((size_t)kN1 * 4);
    bf16_t* w1t    = (bf16_t*)take((size_t)kN1 * kD * 2);   // (N1 x K) N-major
    bf16_t* finwt  = (bf16_t*)take((size_t)kD * kD * 2);    // (N x K) N-major
    bf16_t* fc0wt  = (bf16_t*)take((size_t)kD * kD * 2);
    bf16_t* fc1wt  = (bf16_t*)take((size_t)kD * kD * 2);
    (void)ws_size; (void)in_sizes; (void)n_in; (void)out_size;

    init_x_k<<<(kB * kD + 255) / 256, 256, 0, stream>>>(x, x16);
    prep_kernel<<<kB, kP, 0, stream>>>(obs, bfb, posb);

    for (int l = 0; l < kL; ++l) {
        const float* kvbw_l = kvb_w + (size_t)l * kDPREP * 1024;

        bias1_k<<<(kN1 + 255) / 256, 256, 0, stream>>>(
            kvx_b + (size_t)l * 1024, kvb_b + (size_t)l * 1024, q_b + (size_t)l * 512, bias1);

        // per-layer weight transposes (f32 -> bf16, N-major for the GEMM B side)
        tpose_bf16_k<<<dim3(1024 / 32, kD / 32), 256, 0, stream>>>(
            kvx_w + (size_t)l * kD * 1024, w1t, kD, 1024);                 // rows 0..1023
        tpose_bf16_k<<<dim3(512 / 32, kD / 32), 256, 0, stream>>>(
            q_w + (size_t)l * kD * 512, w1t + (size_t)1024 * kD, kD, 512); // rows 1024..1535
        tpose_bf16_k<<<dim3(kD / 32, kD / 32), 256, 0, stream>>>(
            fin_w + (size_t)l * kD * kD, finwt, kD, kD);
        tpose_bf16_k<<<dim3(kD / 32, kD / 32), 256, 0, stream>>>(
            fc0_w + (size_t)l * kD * kD, fc0wt, kD, kD);
        tpose_bf16_k<<<dim3(kD / 32, kD / 32), 256, 0, stream>>>(
            fc1_w + (size_t)l * kD * kD, fc1wt, kD, kD);

        // fused [kvx | q] projection: (256x512) @ (512x1536) + bias -> out1 (f32)
        gemm_bf16_k<0><<<dim3(kB / 64, kN1 / 64), 128, 0, stream>>>(
            x16, w1t, kB, kN1, kD, bias1, out1, nullptr, nullptr, nullptr);

        // rank-19 attention path
        s_kernel<<<dim3(kDPREP, kB), 64, 0, stream>>>(out1, kvbw_l, sbuf);
        attn_kernel<<<kB, 128, 0, stream>>>(out1, sbuf, bfb, kvbw_l, valsf, vals16);

        // fin: x += alpha * relu(vals @ fin_w + fin_b)
        gemm_bf16_k<1><<<dim3(kB / 64, kD / 64), 128, 0, stream>>>(
            vals16, finwt, kB, kD, kD, fin_b + (size_t)l * kD, nullptr, x, x16, alpha + l);
        // fc0: h = relu(x @ fc0_w + fc0_b)
        gemm_bf16_k<2><<<dim3(kB / 64, kD / 64), 128, 0, stream>>>(
            x16, fc0wt, kB, kD, kD, fc0_b + (size_t)l * kD, nullptr, nullptr, h16, nullptr);
        // fc1: x += alpha * (h @ fc1_w + fc1_b)
        gemm_bf16_k<3><<<dim3(kB / 64, kD / 64), 128, 0, stream>>>(
            h16, fc1wt, kB, kD, kD, fc1_b + (size_t)l * kD, nullptr, x, x16, alpha + l);
    }

    head_kernel<<<kB, 128, 0, stream>>>(x, head_w, posb, out);
}
